// PointNet_SA_Module_1967095021876
// MI455X (gfx1250) — matrix operand order repaired and checked by reading, not compiled, not run on hardware
//
#include <hip/hip_runtime.h>

typedef _Float16 half_t;
typedef __attribute__((ext_vector_type(16))) _Float16 v16h;
typedef __attribute__((ext_vector_type(8)))  _Float16 v8h;
typedef __attribute__((ext_vector_type(8)))  float    v8f;

#define BB   16
#define NN   8192
#define SS   512
#define KK   32
#define C0   67
#define C0P  96
#define H0   64
#define H1   64
#define H2   128
#define RAD2 0.16f
#define EPSV 1e-5f

// Intermediate activations h are stored CHANNEL-MAJOR: h[g][c][m] (m = 0..31).
// A lane's 8 C-fragment values are contiguous in m -> one 32B store.

// ---------------------------------------------------------------- FPS
__global__ void fps_kernel(const float* __restrict__ xyz, int* __restrict__ fidx) {
  const int b   = blockIdx.x;
  const int tid = threadIdx.x;                 // 1024 threads
  const float* xp = xyz + (size_t)b * 3 * NN;
  const float* yp = xp + NN;
  const float* zp = yp + NN;
  const int PPT = NN / 1024;                   // 8 points per thread
  float dist[8];
#pragma unroll
  for (int i = 0; i < PPT; ++i) dist[i] = 3.4e38f;

  __shared__ float rv[1024];
  __shared__ int   ri[1024];
  __shared__ int   s_last;

  int last = 0;
  for (int s = 0; s < SS; ++s) {
    if (tid == 0) fidx[b * SS + s] = last;
    const float lx = xp[last], ly = yp[last], lz = zp[last];
    float bestv = -1.0f;
    int   besti = 0;
#pragma unroll
    for (int i = 0; i < PPT; ++i) {
      const int n = i * 1024 + tid;
      const float dx = xp[n] - lx, dy = yp[n] - ly, dz = zp[n] - lz;
      float d = dx * dx + dy * dy + dz * dz;
      d = fminf(dist[i], d);
      dist[i] = d;
      if (d > bestv) { bestv = d; besti = n; }  // ascending n -> first max kept
    }
    rv[tid] = bestv; ri[tid] = besti;
    __syncthreads();
    for (int off = 512; off > 0; off >>= 1) {
      if (tid < off) {
        const float v2 = rv[tid + off]; const int i2 = ri[tid + off];
        if (v2 > rv[tid] || (v2 == rv[tid] && i2 < ri[tid])) { rv[tid] = v2; ri[tid] = i2; }
      }
      __syncthreads();
    }
    if (tid == 0) s_last = ri[0];
    __syncthreads();
    last = s_last;
  }
}

// ---------------------------------------------------------------- new_xyz (B,3,S)
__global__ void newxyz_kernel(const float* __restrict__ xyz, const int* __restrict__ fidx,
                              float* __restrict__ out) {
  const int b = blockIdx.x, s = threadIdx.x;   // 512 threads
  const int f = fidx[b * SS + s];
#pragma unroll
  for (int d = 0; d < 3; ++d)
    out[(size_t)b * 3 * SS + (size_t)d * SS + s] = xyz[(size_t)b * 3 * NN + (size_t)d * NN + f];
}

// ---------------------------------------------------------------- ball query
__global__ void ballq_kernel(const float* __restrict__ xyz, const int* __restrict__ fidx,
                             int* __restrict__ idx) {
  const int gid = blockIdx.x * (blockDim.x >> 5) + (threadIdx.x >> 5);
  if (gid >= BB * SS) return;
  const int lane = threadIdx.x & 31;
  const int b = gid / SS, s = gid % SS;
  const float* xp = xyz + (size_t)b * 3 * NN;
  const float* yp = xp + NN;
  const float* zp = yp + NN;
  const int f = fidx[b * SS + s];
  const float cx = xp[f], cy = yp[f], cz = zp[f];
  int* op = idx + (size_t)gid * KK;

  int filled = 0;
  int first  = f;
  for (int n0 = 0; n0 < NN && filled < KK; n0 += 32) {
    const int n = n0 + lane;
    const float dx = xp[n] - cx, dy = yp[n] - cy, dz = zp[n] - cz;
    const bool in = (dx * dx + dy * dy + dz * dz) < RAD2;
    const unsigned mask = __builtin_amdgcn_ballot_w32(in);
    if (filled == 0 && mask) first = n0 + __builtin_ctz(mask);
    const int prefix = __builtin_popcount(mask & ((1u << lane) - 1u));
    if (in && (filled + prefix) < KK) op[filled + prefix] = n;
    filled += __builtin_popcount(mask);
  }
  for (int t = (filled < KK ? filled : KK) + lane; t < KK; t += 32) op[t] = first;
}

// ---------------------------------------------------------------- WMMA tile helper
// Xs: row-major activations (xstride halves per row, 16B-aligned rows).
// Wt: TRANSPOSED weights, N-major (kstride halves per output channel):
//     each lane's 16 B-fragment halves are contiguous -> ds_load_b128 pairs.
__device__ __forceinline__ v8f wmma_tile(const half_t* __restrict__ Xs, int xstride,
                                         const half_t* __restrict__ Wt, int kstride,
                                         int arow, int ncol, int hi, int ksteps, v8f acc) {
#pragma unroll
  for (int kt = 0; kt < ksteps; ++kt) {
    const int k0 = kt * 32;
    const v8h alo = *(const v8h*)&Xs[arow * xstride + k0 + hi * 16];
    const v8h ahi = *(const v8h*)&Xs[arow * xstride + k0 + hi * 16 + 8];
    const v16h a = __builtin_shufflevector(alo, ahi, 0, 1, 2, 3, 4, 5, 6, 7,
                                           8, 9, 10, 11, 12, 13, 14, 15);
    const v16h w = *(const v16h*)&Wt[ncol * kstride + k0 + hi * 16];
    acc = __builtin_amdgcn_wmma_f32_16x16x32_f16(false, a, false, w, (short)0, acc,
                                                 false, false);
  }
  return acc;
}

// Epilogue: bias-add, contiguous 8-float store into h[g][c][m], sums into LDS.
__device__ __forceinline__ void tile_epilogue(v8f acc, float bias, float* __restrict__ dst,
                                              float* __restrict__ lsum, float* __restrict__ lsq,
                                              int ncol) {
  v8f o;
  float ps = 0.f, ps2 = 0.f;
#pragma unroll
  for (int r = 0; r < 8; ++r) {
    const float v = acc[r] + bias;
    o[r] = v; ps += v; ps2 += v * v;
  }
  *(v8f*)dst = o;                       // 32B contiguous -> 2x global_store_b128
  atomicAdd(&lsum[ncol], ps);
  atomicAdd(&lsq[ncol], ps2);
}

// ---------------------------------------------------------------- layer 0: gather + GEMM 32x67 * 67x64
__global__ void gemm0_kernel(const float* __restrict__ xyz, const float* __restrict__ pts,
                             const int* __restrict__ fidx, const int* __restrict__ idx,
                             const float* __restrict__ w0, const float* __restrict__ b0,
                             float* __restrict__ h0, float* __restrict__ gsum, float* __restrict__ gsq) {
  const int g = blockIdx.x;                    // B*S groups
  const int b = g / SS, s = g % SS;
  const int tid = threadIdx.x;                 // 128 threads
  __shared__ alignas(32) half_t Xs[KK * C0P];  // 32 x 96 row-major
  __shared__ alignas(32) half_t Wt[H0 * C0P];  // 64 x 96 N-major (transposed)
  __shared__ int    sidx[KK];
  __shared__ float  cen[3];
  __shared__ float  lsum[H0], lsq[H0];

  if (tid < KK) sidx[tid] = idx[(size_t)g * KK + tid];
  if (tid < H0) { lsum[tid] = 0.f; lsq[tid] = 0.f; }
  if (tid == 0) {
    const int f = fidx[b * SS + s];
    cen[0] = xyz[(size_t)b * 3 * NN + f];
    cen[1] = xyz[(size_t)b * 3 * NN + NN + f];
    cen[2] = xyz[(size_t)b * 3 * NN + 2 * NN + f];
  }
  __syncthreads();

  for (int e = tid; e < KK * C0P; e += 128) {
    const int row = e / C0P, col = e % C0P;
    float v = 0.f;
    const int p = sidx[row];
    if (col < 3)        v = xyz[(size_t)b * 3 * NN + (size_t)col * NN + p] - cen[col];
    else if (col < C0)  v = pts[(size_t)b * 64 * NN + (size_t)(col - 3) * NN + p];
    Xs[e] = (half_t)v;
  }
  for (int e = tid; e < H0 * C0P; e += 128) {
    const int n = e / C0P, k = e % C0P;
    Wt[e] = (half_t)((k < C0) ? w0[k * H0 + n] : 0.f);
  }
  __syncthreads();

  const int wv = tid >> 5, lane = tid & 31;
  const int ln = lane & 15, hi = lane >> 4;
  const int ncol = wv * 16 + ln;
  const float bias = b0[ncol];
  for (int mt = 0; mt < 2; ++mt) {
    v8f acc = {};
    acc = wmma_tile(Xs, C0P, Wt, C0P, mt * 16 + ln, ncol, hi, C0P / 32, acc);
    tile_epilogue(acc, bias,
                  &h0[((size_t)g * H0 + ncol) * KK + mt * 16 + hi * 8],
                  lsum, lsq, ncol);
  }
  __syncthreads();
  if (tid < H0) { atomicAdd(&gsum[tid], lsum[tid]); atomicAdd(&gsq[tid], lsq[tid]); }
}

// ---------------------------------------------------------------- BN stats -> scale/shift
__global__ void stats_kernel(const float* __restrict__ gsum, const float* __restrict__ gsq,
                             const float* __restrict__ gamma, const float* __restrict__ beta,
                             float* __restrict__ scale, float* __restrict__ shift, int C) {
  const int c = threadIdx.x;
  if (c < C) {
    const float n = (float)(BB * SS * KK);
    const float mean = gsum[c] / n;
    const float var  = gsq[c] / n - mean * mean;
    const float sc   = gamma[c] * rsqrtf(var + EPSV);
    scale[c] = sc;
    shift[c] = beta[c] - mean * sc;
  }
}

// ---------------------------------------------------------------- layer 1: BN0+ReLU on load, 32x64 * 64x64
__global__ void gemm1_kernel(const float* __restrict__ hin, const float* __restrict__ w1,
                             const float* __restrict__ b1,
                             const float* __restrict__ scale, const float* __restrict__ shift,
                             float* __restrict__ hout, float* __restrict__ gsum, float* __restrict__ gsq) {
  const int g = blockIdx.x;
  const int tid = threadIdx.x;
  __shared__ alignas(32) half_t Xs[KK * H0];   // 32 x 64 row-major (m rows, c cols)
  __shared__ alignas(32) half_t Wt[H1 * H0];   // 64 x 64 N-major (transposed)
  __shared__ float  lsum[H1], lsq[H1];
  if (tid < H1) { lsum[tid] = 0.f; lsq[tid] = 0.f; }
  // hin layout (g, c, m): e = c*KK + m; coalesced global read, scattered LDS write
  for (int e = tid; e < KK * H0; e += 128) {
    const int c = e / KK, m = e % KK;
    float v = hin[(size_t)g * KK * H0 + e];
    v = fmaxf(v * scale[c] + shift[c], 0.f);
    Xs[m * H0 + c] = (half_t)v;
  }
  for (int e = tid; e < H1 * H0; e += 128) {
    const int n = e / H0, k = e % H0;
    Wt[e] = (half_t)w1[k * H1 + n];
  }
  __syncthreads();

  const int wv = tid >> 5, lane = tid & 31;
  const int ln = lane & 15, hi = lane >> 4;
  const int ncol = wv * 16 + ln;
  const float bias = b1[ncol];
  for (int mt = 0; mt < 2; ++mt) {
    v8f acc = {};
    acc = wmma_tile(Xs, H0, Wt, H0, mt * 16 + ln, ncol, hi, H0 / 32, acc);
    tile_epilogue(acc, bias,
                  &hout[((size_t)g * H1 + ncol) * KK + mt * 16 + hi * 8],
                  lsum, lsq, ncol);
  }
  __syncthreads();
  if (tid < H1) { atomicAdd(&gsum[tid], lsum[tid]); atomicAdd(&gsq[tid], lsq[tid]); }
}

// ---------------------------------------------------------------- layer 2: BN1+ReLU on load, 32x64 * 64x128
__global__ void gemm2_kernel(const float* __restrict__ hin, const float* __restrict__ w2,
                             const float* __restrict__ b2,
                             const float* __restrict__ scale, const float* __restrict__ shift,
                             float* __restrict__ hout, float* __restrict__ gsum, float* __restrict__ gsq) {
  const int g = blockIdx.x;
  const int tid = threadIdx.x;
  __shared__ alignas(32) half_t Xs[KK * H1];   // 32 x 64 row-major (m rows, c cols)
  __shared__ alignas(32) half_t Wt[H2 * H1];   // 128 x 64 N-major (transposed)
  __shared__ float  lsum[H2], lsq[H2];
  if (tid < H2) { lsum[tid] = 0.f; lsq[tid] = 0.f; }
  for (int e = tid; e < KK * H1; e += 128) {
    const int c = e / KK, m = e % KK;
    float v = hin[(size_t)g * KK * H1 + e];
    v = fmaxf(v * scale[c] + shift[c], 0.f);
    Xs[m * H1 + c] = (half_t)v;
  }
  for (int e = tid; e < H2 * H1; e += 128) {
    const int n = e / H1, k = e % H1;
    Wt[e] = (half_t)w2[k * H2 + n];
  }
  __syncthreads();

  const int wv = tid >> 5, lane = tid & 31;
  const int ln = lane & 15, hi = lane >> 4;
  for (int nt = wv; nt < 8; nt += 4) {          // 8 N-tiles, 4 waves -> 2 each
    const int ncol = nt * 16 + ln;
    const float bias = b2[ncol];
    for (int mt = 0; mt < 2; ++mt) {
      v8f acc = {};
      acc = wmma_tile(Xs, H1, Wt, H1, mt * 16 + ln, ncol, hi, H1 / 32, acc);
      tile_epilogue(acc, bias,
                    &hout[((size_t)g * H2 + ncol) * KK + mt * 16 + hi * 8],
                    lsum, lsq, ncol);
    }
  }
  __syncthreads();
  if (tid < H2) { atomicAdd(&gsum[tid], lsum[tid]); atomicAdd(&gsq[tid], lsq[tid]); }
}

// ---------------------------------------------------------------- BN2+ReLU + max over K, write (B,128,S)
__global__ void finalize_kernel(const float* __restrict__ h2,
                                const float* __restrict__ scale, const float* __restrict__ shift,
                                float* __restrict__ outp) {
  const int g = blockIdx.x, c = threadIdx.x;   // 128 threads
  const int b = g / SS, s = g % SS;
  const float sc = scale[c], sh = shift[c];
  const float* hp = h2 + ((size_t)g * H2 + c) * KK;   // 32 contiguous floats
  float m = -3.4e38f;
#pragma unroll
  for (int k = 0; k < KK; ++k) {
    float v = hp[k] * sc + sh;
    v = fmaxf(v, 0.f);
    m = fmaxf(m, v);
  }
  outp[(size_t)b * H2 * SS + (size_t)c * SS + s] = m;
}

// ---------------------------------------------------------------- zero helper
__global__ void zero_kernel(float* __restrict__ p, int n) {
  const int i = blockIdx.x * blockDim.x + threadIdx.x;
  if (i < n) p[i] = 0.f;
}

// ---------------------------------------------------------------- launch
extern "C" void kernel_launch(void* const* d_in, const int* in_sizes, int n_in,
                              void* d_out, int out_size, void* d_ws, size_t ws_size,
                              hipStream_t stream) {
  (void)in_sizes; (void)n_in; (void)out_size; (void)ws_size;
  const float* xyz = (const float*)d_in[0];
  const float* pts = (const float*)d_in[1];
  const float* w0 = (const float*)d_in[2];  const float* b0 = (const float*)d_in[3];
  const float* g0 = (const float*)d_in[4];  const float* t0 = (const float*)d_in[5];
  const float* w1 = (const float*)d_in[6];  const float* b1 = (const float*)d_in[7];
  const float* g1 = (const float*)d_in[8];  const float* t1 = (const float*)d_in[9];
  const float* w2 = (const float*)d_in[10]; const float* b2 = (const float*)d_in[11];
  const float* g2 = (const float*)d_in[12]; const float* t2 = (const float*)d_in[13];

  char* ws = (char*)d_ws;
  int*   fidx = (int*)ws;                                  // B*S
  int*   bidx = (int*)(ws + (size_t)BB * SS * 4);          // B*S*K
  float* sums = (float*)(ws + (size_t)BB * SS * 4 + (size_t)BB * SS * KK * 4);
  float* sum0 = sums;        float* sq0 = sums + 64;
  float* sum1 = sums + 128;  float* sq1 = sums + 192;
  float* sum2 = sums + 256;  float* sq2 = sums + 384;      // 512 floats total
  float* bn   = sums + 512;
  float* sc0 = bn;        float* sh0 = bn + 64;
  float* sc1 = bn + 128;  float* sh1 = bn + 192;
  float* sc2 = bn + 256;  float* sh2 = bn + 384;           // 512 floats total
  // big buffers, 4MB-aligned region: h0/h2 share (h0 dead after gemm1)
  char* big = ws + (4u << 20);
  float* hA = (float*)big;                                 // h0 (64MB) then h2 (128MB)
  float* hB = (float*)(big + ((size_t)128 << 20));         // h1 (64MB)

  float* out_xyz = (float*)d_out;                          // (B,3,S)
  float* out_pts = out_xyz + (size_t)BB * 3 * SS;          // (B,128,S)

  zero_kernel<<<1, 512, 0, stream>>>(sums, 512);
  fps_kernel<<<BB, 1024, 0, stream>>>(xyz, fidx);
  newxyz_kernel<<<BB, SS, 0, stream>>>(xyz, fidx, out_xyz);
  ballq_kernel<<<(BB * SS) / 8, 256, 0, stream>>>(xyz, fidx, bidx);

  gemm0_kernel<<<BB * SS, 128, 0, stream>>>(xyz, pts, fidx, bidx, w0, b0, hA, sum0, sq0);
  stats_kernel<<<1, 64, 0, stream>>>(sum0, sq0, g0, t0, sc0, sh0, 64);

  gemm1_kernel<<<BB * SS, 128, 0, stream>>>(hA, w1, b1, sc0, sh0, hB, sum1, sq1);
  stats_kernel<<<1, 64, 0, stream>>>(sum1, sq1, g1, t1, sc1, sh1, 64);

  gemm2_kernel<<<BB * SS, 128, 0, stream>>>(hB, w2, b2, sc1, sh1, hA, sum2, sq2);
  stats_kernel<<<1, 128, 0, stream>>>(sum2, sq2, g2, t2, sc2, sh2, 128);

  finalize_kernel<<<BB * SS, 128, 0, stream>>>(hA, sc2, sh2, out_pts);
}
